// MultiHeadAttention_17892833755600
// MI455X (gfx1250) — matrix operand order repaired and checked by reading, not compiled, not run on hardware
//
#include <hip/hip_runtime.h>
#include <hip/hip_bf16.h>

// Toggle for the Tensor-Data-Mover staging path (fallback = direct global loads)
#define USE_TDM 1

// ---------------------------------------------------------------------------
// Problem constants (from reference): B=2, S=2048, D=1024, H=16, DK=64
// ---------------------------------------------------------------------------
#define BATCH 2
#define SEQ   2048
#define DIM   1024
#define HEADS 16
#define DKH   64          // DIM / HEADS
#define MTOT  (BATCH*SEQ) // 4096 rows in all projection GEMMs

typedef __bf16 v16bf __attribute__((ext_vector_type(16)));
typedef __bf16 v8bf  __attribute__((ext_vector_type(8)));
typedef float  v8f   __attribute__((ext_vector_type(8)));
typedef float  v4f   __attribute__((ext_vector_type(4)));
typedef unsigned int u32x4 __attribute__((ext_vector_type(4)));
typedef unsigned int u32x8 __attribute__((ext_vector_type(8)));

static __device__ __forceinline__ __bf16 f2bf(float f) { return (__bf16)f; }

static __device__ __forceinline__ v8f wmma_bf16(v16bf a, v16bf b, v8f c) {
  // D = A(16x32 bf16) x B(32x16 bf16) + C(16x16 f32)
  return __builtin_amdgcn_wmma_f32_16x16x32_bf16(
      /*neg_a=*/false, a, /*neg_b=*/false, b,
      /*c_mod=*/(short)0, c, /*reuse_a=*/false, /*reuse_b=*/false);
}

static __device__ __forceinline__ v8f v8f_zero() {
  v8f z = {0.f,0.f,0.f,0.f,0.f,0.f,0.f,0.f};
  return z;
}

// ---------------------------------------------------------------------------
// Tensor Data Mover: 2D tile (tile0 contiguous elems per row, tile1 rows,
// row stride stride0, bf16 elements) from global -> LDS, packed rows in LDS.
// D# built per CDNA5 ISA; tracked by TENSORcnt.
// ---------------------------------------------------------------------------
#if USE_TDM
static __device__ __forceinline__ void tdm_load_to_lds(
    unsigned lds_off, const void* gptr,
    unsigned tile0, unsigned tile1, unsigned stride0) {
  const unsigned long long ga = (unsigned long long)(uintptr_t)gptr;
  u32x4 g0;
  g0[0] = 1u;                                   // count=1 (valid), user mode
  g0[1] = lds_off;                              // lds_addr (bytes)
  g0[2] = (unsigned)ga;                         // global_addr[31:0]
  g0[3] = (unsigned)((ga >> 32) & 0x01FFFFFFu)  // global_addr[56:32]
        | (2u << 30);                           // type = 2 ("image")
  const unsigned TD = 0x100000u;                // huge tensor dims: no OOB clip
  u32x8 g1;
  g1[0] = 1u << 16;                             // wg_mask=0, data_size=1 (2B)
  g1[1] = (TD & 0xFFFFu) << 16;                 // tensor_dim0[15:0]
  g1[2] = ((TD >> 16) & 0xFFFFu)                // tensor_dim0[31:16]
        | ((TD & 0xFFFFu) << 16);               // tensor_dim1[15:0]
  g1[3] = ((TD >> 16) & 0xFFFFu)                // tensor_dim1[31:16]
        | ((tile0 & 0xFFFFu) << 16);            // tile_dim0
  g1[4] = (tile1 & 0xFFFFu);                    // tile_dim1 (tile_dim2 = 0)
  g1[5] = stride0;                              // tensor_dim0_stride[31:0]
  g1[6] = 0u;                                   // stride0 hi / stride1 lo
  g1[7] = 0u;                                   // stride1 hi
  asm volatile("tensor_load_to_lds %0, %1" :: "s"(g0), "s"(g1) : "memory");
}
static __device__ __forceinline__ unsigned lds_off_of(const void* p) {
  return (unsigned)(uintptr_t)p;                // low 32 bits of LDS aperture
}
#endif

// ---------------------------------------------------------------------------
// Cross-lane xor butterfly within 16-lane groups via v_permlane16_b32.
// ---------------------------------------------------------------------------
#if __has_builtin(__builtin_amdgcn_permlane16)
#define XOR16(v, s0, s1, m)                                                   \
  __int_as_float(__builtin_amdgcn_permlane16(                                 \
      __float_as_int(v), __float_as_int(v), (int)(s0), (int)(s1), false, false))
#else
#define XOR16(v, s0, s1, m) __shfl_xor((v), (m), 32)
#endif

static __device__ __forceinline__ float red_max16(float v) {
  v = fmaxf(v, XOR16(v, 0x67452301u, 0xEFCDAB89u, 1));   // xor 1
  v = fmaxf(v, XOR16(v, 0x54761032u, 0xDCFE98BAu, 2));   // xor 2
  v = fmaxf(v, XOR16(v, 0x32107654u, 0xBA98FEDCu, 4));   // xor 4
  v = fmaxf(v, XOR16(v, 0xFEDCBA98u, 0x76543210u, 8));   // xor 8
  return v;
}
static __device__ __forceinline__ float red_sum16(float v) {
  v += XOR16(v, 0x67452301u, 0xEFCDAB89u, 1);
  v += XOR16(v, 0x54761032u, 0xDCFE98BAu, 2);
  v += XOR16(v, 0x32107654u, 0xBA98FEDCu, 4);
  v += XOR16(v, 0xFEDCBA98u, 0x76543210u, 8);
  return v;
}

// 16-bit A-fragment (16x32, MxK): elems e -> K = k0 + g*16 + e (matches B fill).
static __device__ __forceinline__ v16bf make_afrag(const __bf16* rowp, int k0, int g) {
  const v8bf lo = *(const v8bf*)(rowp + k0 + g * 16);
  const v8bf hi = *(const v8bf*)(rowp + k0 + g * 16 + 8);
  v16bf a;
#pragma unroll
  for (int e = 0; e < 8; ++e) { a[e] = lo[e]; a[8 + e] = hi[e]; }
  return a;
}

// ---------------------------------------------------------------------------
// 1) fp32 -> bf16 conversion (vectorized, optional scale folded in)
// ---------------------------------------------------------------------------
__global__ void cvt_f32_bf16(const float* __restrict__ in,
                             __bf16* __restrict__ out, int n8, float scale) {
  int i = blockIdx.x * blockDim.x + threadIdx.x;
  if (i < n8) {
    const v4f* p = (const v4f*)(in + (size_t)i * 8);
    const v4f x0 = p[0], x1 = p[1];
    v8bf y;
#pragma unroll
    for (int e = 0; e < 4; ++e) {
      y[e]     = f2bf(x0[e] * scale);
      y[4 + e] = f2bf(x1[e] * scale);
    }
    *(v8bf*)(out + (size_t)i * 8) = y;
  }
}

// ---------------------------------------------------------------------------
// 2) Projection GEMM:  Y[M,N] = X[M,K] * W[N,K]^T   (M=4096, N=K=1024)
//    MODE 0: bf16 head-major [B,H,S,DK]; MODE 1: bf16 [B,H,DK,S]; MODE 2: f32.
//    Block = 4 waves; all waves share one 64x64 W K-slab staged by TDM into
//    LDS (double-buffered); A rows stream from global (wave-private).
// ---------------------------------------------------------------------------
template <int MODE>
__global__ __launch_bounds__(128)
void gemm_bf16(const __bf16* __restrict__ X, const __bf16* __restrict__ W,
               __bf16* __restrict__ outb, float* __restrict__ outf) {
  const int lane = threadIdx.x & 31;
  const int wave = threadIdx.x >> 5;
  const int g = lane >> 4, ln = lane & 15;

  const int m0 = blockIdx.x * 64 + wave * 16;
  const int n0 = blockIdx.y * 64;

#if USE_TDM
  __shared__ alignas(16) __bf16 wbuf[2][64 * 64];   // 2 x 8KB W tiles
#endif

  v8f acc[4];
#pragma unroll
  for (int nb = 0; nb < 4; ++nb) acc[nb] = v8f_zero();

  const __bf16* xrow = X + (size_t)(m0 + ln) * DIM;  // A row (M = m0+ln)

#if USE_TDM
  if (wave == 0)    // prologue: DMA first W tile (64 rows x 64 K, stride DIM)
    tdm_load_to_lds(lds_off_of(&wbuf[0][0]), W + (size_t)n0 * DIM, 64, 64, DIM);
  int ib = 0;
#endif

  for (int k0 = 0; k0 < DIM; k0 += 64) {
    if (k0 + 64 < DIM) __builtin_prefetch(xrow + k0 + 64, 0, 3);
#if USE_TDM
    if (wave == 0) __builtin_amdgcn_s_wait_tensorcnt(0);
    __syncthreads();                             // tile ib ready for all waves
    if (wave == 0 && k0 + 64 < DIM)
      tdm_load_to_lds(lds_off_of(&wbuf[ib ^ 1][0]),
                      W + (size_t)n0 * DIM + k0 + 64, 64, 64, DIM);
#endif
    const v16bf a0 = make_afrag(xrow, k0, g);
    const v16bf a1 = make_afrag(xrow, k0 + 32, g);
#pragma unroll
    for (int nb = 0; nb < 4; ++nb) {
      // B-fragment (32x16): lane = col N, elems e -> K = k0 + g*16 + e.
#if USE_TDM
      const __bf16* wrow = &wbuf[ib][(nb * 16 + ln) * 64];
#else
      const __bf16* wrow = W + (size_t)(n0 + nb * 16 + ln) * DIM + k0;
#endif
      const v16bf b0 = *(const v16bf*)(wrow + g * 16);
      const v16bf b1 = *(const v16bf*)(wrow + 32 + g * 16);
      acc[nb] = wmma_bf16(a0, b0, acc[nb]);
      acc[nb] = wmma_bf16(a1, b1, acc[nb]);
    }
#if USE_TDM
    ib ^= 1;
#endif
  }

  // Store. C layout: VGPR r -> (M = m0 + r + 8g, N = n0 + nb*16 + ln)
#pragma unroll
  for (int nb = 0; nb < 4; ++nb) {
    const int ncol = n0 + nb * 16 + ln;
#pragma unroll
    for (int r = 0; r < 8; ++r) {
      const int row = m0 + r + 8 * g;          // global token index
      const float val = acc[nb][r];
      if (MODE == 2) {
        outf[(size_t)row * DIM + ncol] = val;
      } else {
        const int b = row >> 11;               // / SEQ
        const int s = row & (SEQ - 1);
        const int h = ncol >> 6;               // / DKH
        const int dk = ncol & (DKH - 1);
        size_t idx;
        if (MODE == 0) idx = (((size_t)(b * HEADS + h)) * SEQ + s) * DKH + dk;
        else           idx = (((size_t)(b * HEADS + h)) * DKH + dk) * SEQ + s;
        outb[idx] = f2bf(val);
      }
    }
  }
}

// ---------------------------------------------------------------------------
// 3) Flash attention, 64-key tiles, online softmax in log2 domain.
//    Q pre-scaled by (1/sqrt(DK))*log2(e); v_exp_f32 computes exp2 natively.
//    K/V tiles (8KB each) are TDM-DMA'd into double-buffered LDS and shared
//    by all 4 waves (each wave owns one 16-query tile).
// ---------------------------------------------------------------------------
__global__ __launch_bounds__(128)
void attn_flash(const __bf16* __restrict__ qh, const __bf16* __restrict__ kh,
                const __bf16* __restrict__ vT, __bf16* __restrict__ ctx) {
  const int lane = threadIdx.x & 31;
  const int wave = threadIdx.x >> 5;
  const int g = lane >> 4, ln = lane & 15;

  const int bh = blockIdx.x >> 5;       // b*HEADS + h
  const int qt = blockIdx.x & 31;
  const int q0 = qt * 64 + wave * 16;

  __shared__ alignas(16) __bf16 pbuf[4][16 * 64];   // per-wave P tiles (8KB)
#if USE_TDM
  __shared__ alignas(16) __bf16 kbuf[2][64 * 64];   // K tiles: row=key, 64 dk
  __shared__ alignas(16) __bf16 vbuf[2][64 * 64];   // V tiles: row=dk, 64 keys
#endif
  __bf16* P = &pbuf[wave][0];

  const __bf16* Qb = qh + (size_t)bh * SEQ * DKH;
  const __bf16* Kb = kh + (size_t)bh * SEQ * DKH;
  const __bf16* Vb = vT + (size_t)bh * DKH * SEQ;

  // Q fragments: 16 queries x DK=64 -> two A-frags (K-splits 0..31, 32..63)
  v16bf aq0, aq1;
  {
    const __bf16* qrow = Qb + (size_t)(q0 + ln) * DKH;
    aq0 = make_afrag(qrow, 0, g);
    aq1 = make_afrag(qrow, 32, g);
  }

  v8f acc[4];
#pragma unroll
  for (int nb = 0; nb < 4; ++nb) acc[nb] = v8f_zero();
  float mrow[8], lrow[8];
#pragma unroll
  for (int r = 0; r < 8; ++r) { mrow[r] = -3.0e38f; lrow[r] = 0.f; }

#if USE_TDM
  if (wave == 0) {   // prologue: DMA first K (contiguous 8KB) and V (2D) tiles
    tdm_load_to_lds(lds_off_of(&kbuf[0][0]), Kb, 4096, 1, 4096);
    tdm_load_to_lds(lds_off_of(&vbuf[0][0]), Vb, 64, 64, SEQ);
  }
  int ib = 0;
#endif

  for (int kb = 0; kb < SEQ; kb += 64) {
#if USE_TDM
    if (wave == 0) __builtin_amdgcn_s_wait_tensorcnt(0);
    __syncthreads();                       // tiles ib ready; prev reads done
    if (wave == 0 && kb + 64 < SEQ) {
      tdm_load_to_lds(lds_off_of(&kbuf[ib ^ 1][0]),
                      Kb + (size_t)(kb + 64) * DKH, 4096, 1, 4096);
      tdm_load_to_lds(lds_off_of(&vbuf[ib ^ 1][0]),
                      Vb + kb + 64, 64, 64, SEQ);
    }
    const __bf16* Kt = &kbuf[ib][0];       // row stride 64 (dk-contiguous)
    const __bf16* Vt = &vbuf[ib][0];       // row stride 64 (key-contiguous)
    const int VSTRIDE = 64;
#else
    const __bf16* Kt = Kb + (size_t)kb * DKH;
    const __bf16* Vt = Vb + kb;
    const int VSTRIDE = SEQ;
#endif

    // ---- scores for 64 keys: four 16x16 f32 tiles (8 WMMA) ----
    v8f s[4];
#pragma unroll
    for (int t = 0; t < 4; ++t) {
      const __bf16* krow = Kt + (size_t)(t * 16 + ln) * 64;  // col N = key
      const v16bf b0 = *(const v16bf*)(krow + g * 16);        // dk 0..31
      const v16bf b1 = *(const v16bf*)(krow + 32 + g * 16);   // dk 32..63
      v8f z = v8f_zero();
      z = wmma_bf16(aq0, b0, z);
      z = wmma_bf16(aq1, b1, z);
      s[t] = z;
    }

    // ---- online softmax (rows M = r + 8g live on the 16 lanes of group g)
#pragma unroll
    for (int r = 0; r < 8; ++r) {
      const float a0 = s[0][r], a1 = s[1][r], a2 = s[2][r], a3 = s[3][r];
      float mx = fmaxf(fmaxf(a0, a1), fmaxf(a2, a3));
      mx = red_max16(mx);
      const float nm = fmaxf(mrow[r], mx);
      const float corr = exp2f(mrow[r] - nm);
      const float p0 = exp2f(a0 - nm);
      const float p1 = exp2f(a1 - nm);
      const float p2 = exp2f(a2 - nm);
      const float p3 = exp2f(a3 - nm);
      const float rs = red_sum16((p0 + p1) + (p2 + p3));
      lrow[r] = lrow[r] * corr + rs;
      mrow[r] = nm;
#pragma unroll
      for (int nb = 0; nb < 4; ++nb) acc[nb][r] *= corr;
      // stage P tile (row-major 16x64) in LDS for the layout transpose
      __bf16* prow = P + (r + 8 * g) * 64 + ln;
      prow[0]  = f2bf(p0);
      prow[16] = f2bf(p1);
      prow[32] = f2bf(p2);
      prow[48] = f2bf(p3);
    }

    __builtin_amdgcn_wave_barrier();
    asm volatile("s_wait_dscnt 0x0" ::: "memory");  // RAW: P stores visible

    // re-read P as two A-fragments (16 queries x 64 keys)
    const __bf16* prow = P + ln * 64;
    const v16bf ap0 = make_afrag(prow, 0, g);
    const v16bf ap1 = make_afrag(prow, 32, g);
    __builtin_amdgcn_wave_barrier();

    // ---- ctx += P @ V  (contract over 64 keys, N = 4 blocks of DK cols)
#pragma unroll
    for (int nb = 0; nb < 4; ++nb) {
      const __bf16* vrow = Vt + (size_t)(nb * 16 + ln) * VSTRIDE;
      const v16bf bv0 = *(const v16bf*)(vrow + g * 16);
      const v16bf bv1 = *(const v16bf*)(vrow + 32 + g * 16);
      acc[nb] = wmma_bf16(ap0, bv0, acc[nb]);
      acc[nb] = wmma_bf16(ap1, bv1, acc[nb]);
    }
#if USE_TDM
    ib ^= 1;
#endif
  }

  // ---- epilogue: divide by l, store ctx bf16 row-major [token, D] ----
  const int b = bh >> 4;       // / HEADS
  const int h = bh & 15;
#pragma unroll
  for (int r = 0; r < 8; ++r) {
    const float inv = 1.0f / lrow[r];
    const int token = b * SEQ + q0 + r + 8 * g;
#pragma unroll
    for (int nb = 0; nb < 4; ++nb) {
      const int col = h * DKH + nb * 16 + ln;
      ctx[(size_t)token * DIM + col] = f2bf(acc[nb][r] * inv);
    }
  }
}

// ---------------------------------------------------------------------------
// Host launcher
// ---------------------------------------------------------------------------
extern "C" void kernel_launch(void* const* d_in, const int* in_sizes, int n_in,
                              void* d_out, int out_size, void* d_ws, size_t ws_size,
                              hipStream_t stream) {
  const float* q  = (const float*)d_in[0];
  const float* k  = (const float*)d_in[1];
  const float* v  = (const float*)d_in[2];
  const float* Wq = (const float*)d_in[3];
  const float* Wk = (const float*)d_in[4];
  const float* Wv = (const float*)d_in[5];
  const float* Wo = (const float*)d_in[6];
  float* out = (float*)d_out;

  const size_t NX = (size_t)MTOT * DIM;   // 4,194,304
  const size_t NW = (size_t)DIM * DIM;    // 1,048,576

  __bf16* ws  = (__bf16*)d_ws;            // 64 MB of bf16 scratch total
  __bf16* Xq  = ws;
  __bf16* Xk  = Xq + NX;
  __bf16* Xv  = Xk + NX;
  __bf16* Wqb = Xv + NX;
  __bf16* Wkb = Wqb + NW;
  __bf16* Wvb = Wkb + NW;
  __bf16* Wob = Wvb + NW;
  __bf16* qhh = Wob + NW;                 // [B,H,S,DK]
  __bf16* khh = qhh + NX;                 // [B,H,S,DK]
  __bf16* vTT = khh + NX;                 // [B,H,DK,S]
  __bf16* ctx = vTT + NX;                 // [B*S, D]

  // softmax scale folded into Q, in log2 domain: (1/sqrt(DK)) * log2(e)
  const float QSCALE = 0.125f * 1.44269504088896340736f;

  const int CT = 256;
  const int GX = (int)((NX / 8 + CT - 1) / CT);
  const int GW = (int)((NW / 8 + CT - 1) / CT);
  cvt_f32_bf16<<<GX, CT, 0, stream>>>(q, Xq, (int)(NX / 8), QSCALE);
  cvt_f32_bf16<<<GX, CT, 0, stream>>>(k, Xk, (int)(NX / 8), 1.0f);
  cvt_f32_bf16<<<GX, CT, 0, stream>>>(v, Xv, (int)(NX / 8), 1.0f);
  cvt_f32_bf16<<<GW, CT, 0, stream>>>(Wq, Wqb, (int)(NW / 8), 1.0f);
  cvt_f32_bf16<<<GW, CT, 0, stream>>>(Wk, Wkb, (int)(NW / 8), 1.0f);
  cvt_f32_bf16<<<GW, CT, 0, stream>>>(Wv, Wvb, (int)(NW / 8), 1.0f);
  cvt_f32_bf16<<<GW, CT, 0, stream>>>(Wo, Wob, (int)(NW / 8), 1.0f);

  dim3 ggrid(MTOT / 64, DIM / 64);   // 64 x 16
  gemm_bf16<0><<<ggrid, 128, 0, stream>>>(Xq, Wqb, qhh, nullptr);
  gemm_bf16<0><<<ggrid, 128, 0, stream>>>(Xk, Wkb, khh, nullptr);
  gemm_bf16<1><<<ggrid, 128, 0, stream>>>(Xv, Wvb, vTT, nullptr);

  attn_flash<<<BATCH * HEADS * (SEQ / 64), 128, 0, stream>>>(qhh, khh, vTT, ctx);

  gemm_bf16<2><<<ggrid, 128, 0, stream>>>(ctx, Wob, nullptr, out);
}
